// ScaledDotProductAttention_103079215706
// MI455X (gfx1250) — matrix operand order repaired and checked by reading, not compiled, not run on hardware
//
#include <hip/hip_runtime.h>

#define BB 16
#define NN 2048
#define DD 64
/* Q pre-scale: (1/sqrt(2048)) * log2(e)  -> WMMA scores come out in exp2 domain */
#define QSCALE 0.03188227050305353f
#define L2E    1.4426950408889634f

typedef _Float16     h16 __attribute__((ext_vector_type(16)));
typedef float        f8  __attribute__((ext_vector_type(8)));
typedef unsigned int u4  __attribute__((ext_vector_type(4)));

union FragH { h16 h; u4 u[2]; unsigned int w[8]; };

// A-matrix fragment (16 x 32 f16, M rows x K): lanes 0-15: M=li, K=k0+hg*8..+7 and k0+16+hg*8..+7
__device__ __forceinline__ h16 load_frag_a(const _Float16* base, int row0, int ld, int k0, int li, int hg) {
    FragH f;
    const _Float16* p = base + (row0 + li) * ld + k0 + hg * 8;
    f.u[0] = *(const u4*)(p);
    f.u[1] = *(const u4*)(p + 16);
    return f.h;
}

// B-matrix fragment (32 x 16 f16, K x N), storage row-major [N][K]:
// N=li, K=k0+hg*8..+7 and k0+16+hg*8..+7
__device__ __forceinline__ h16 load_frag_b(const _Float16* base, int row0, int ld, int k0, int li, int hg) {
    FragH f;
    const _Float16* p = base + (row0 + li) * ld + k0 + hg * 8;
    f.u[0] = *(const u4*)(p);
    f.u[1] = *(const u4*)(p + 16);
    return f.h;
}

__device__ __forceinline__ f8 wmma_f16(h16 a, h16 b, f8 c) {
    return __builtin_amdgcn_wmma_f32_16x16x32_f16(false, a, false, b, (short)0, c, false, false);
}

// ---- kernel 1: fp32 -> f16 conversion (Q pre-scaled into exp2 domain) ----
__global__ __launch_bounds__(256) void prep_qk(const float* __restrict__ q, const float* __restrict__ k,
                                               _Float16* __restrict__ qh, _Float16* __restrict__ kh) {
    int i = blockIdx.x * blockDim.x + threadIdx.x;
    qh[i] = (_Float16)(q[i] * QSCALE);
    kh[i] = (_Float16)(k[i]);
}

// ---- kernel 2: per-column (axis=1) softmax denominators: iZ[b,m] = 1 / sum_n exp2(c[n,m]) ----
__global__ __launch_bounds__(256) void colsum_kernel(const _Float16* __restrict__ qh,
                                                     const _Float16* __restrict__ kh,
                                                     const unsigned char* __restrict__ mask,
                                                     float* __restrict__ iZ) {
    int wid  = blockIdx.x * (blockDim.x >> 5) + (threadIdx.x >> 5);
    int lane = threadIdx.x & 31;
    int li = lane & 15, hg = lane >> 4;
    int b  = wid >> 7;            // 128 m-tiles per batch
    int m0 = (wid & 127) << 4;
    const _Float16* Kb = kh + b * NN * DD;
    const _Float16* Qb = qh + b * NN * DD;
    const unsigned char* Mb = mask + (size_t)b * NN * NN;

    h16 aK0 = load_frag_a(Kb, m0, DD, 0,  li, hg);   // K-tile fixed for this wave
    h16 aK1 = load_frag_a(Kb, m0, DD, 32, li, hg);

    float z[8];
    #pragma unroll
    for (int r = 0; r < 8; ++r) z[r] = 0.f;

    for (int it = 0; it < NN / 16; ++it) {
        int n0 = it << 4;
        h16 bq0 = load_frag_b(Qb, n0, DD, 0,  li, hg);
        h16 bq1 = load_frag_b(Qb, n0, DD, 32, li, hg);
        f8 c = {0.f,0.f,0.f,0.f,0.f,0.f,0.f,0.f};
        c = wmma_f16(aK0, bq0, c);                    // S^T tile: lane = n, row r = m
        c = wmma_f16(aK1, bq1, c);
        unsigned long long mb =
            *(const unsigned long long*)(Mb + (size_t)(n0 + li) * NN + m0 + hg * 8);
        unsigned mlo = (unsigned)mb;
        unsigned mhi = (unsigned)(mb >> 32);
        #pragma unroll
        for (int r = 0; r < 8; ++r) {
            unsigned byte = ((r < 4 ? mlo : mhi) >> (8 * (r & 3))) & 0xffu;
            float e = __builtin_amdgcn_exp2f((float)c[r]);   // raw v_exp_f32, branchless
            z[r] += byte ? 0.f : e;
        }
    }
    // each 16-lane group holds disjoint n-residues -> butterfly reduce within group
    #pragma unroll
    for (int r = 0; r < 8; ++r) {
        float v = z[r];
        v += __shfl_xor(v, 1, 32);
        v += __shfl_xor(v, 2, 32);
        v += __shfl_xor(v, 4, 32);
        v += __shfl_xor(v, 8, 32);
        z[r] = v;
    }
    if (li == 0) {
        #pragma unroll
        for (int r = 0; r < 8; ++r) {
            float v = z[r];
            iZ[b * NN + m0 + hg * 8 + r] = (v > 0.f) ? 1.f / v : 0.f;
        }
    }
}

// ---- kernel 3: Vt[b][d][m] = f16( V[b][m][d] * iZ[b][m] ), LDS-tiled transpose ----
__global__ __launch_bounds__(256) void prep_vt(const float* __restrict__ v, const float* __restrict__ iZ,
                                               _Float16* __restrict__ vt) {
    __shared__ float tile[32][33];
    int b  = blockIdx.z;
    int m0 = blockIdx.x * 32;
    int d0 = blockIdx.y * 32;
    int tx = threadIdx.x, ty = threadIdx.y;   // 32 x 8
    #pragma unroll
    for (int j = 0; j < 4; ++j) {
        int m = m0 + ty + j * 8;
        tile[ty + j * 8][tx] = v[((size_t)b * NN + m) * DD + d0 + tx] * iZ[b * NN + m];
    }
    __syncthreads();
    #pragma unroll
    for (int j = 0; j < 4; ++j) {
        int d = d0 + ty + j * 8;
        vt[((size_t)b * DD + d) * NN + m0 + tx] = (_Float16)tile[tx][ty + j * 8];
    }
}

// ---- kernel 4: out[n,:] = sum_m exp2(c[n,m]) * Vt_scaled[:,m]  (S^T recompute + PV, fused) ----
__global__ __launch_bounds__(256) void attn_out_kernel(const _Float16* __restrict__ qh,
                                                       const _Float16* __restrict__ kh,
                                                       const _Float16* __restrict__ vt,
                                                       const unsigned char* __restrict__ mask,
                                                       float* __restrict__ out) {
    int wid  = blockIdx.x * (blockDim.x >> 5) + (threadIdx.x >> 5);
    int lane = threadIdx.x & 31;
    int li = lane & 15, hg = lane >> 4;
    int b  = wid >> 7;
    int n0 = (wid & 127) << 4;
    const _Float16* Kb  = kh + b * NN * DD;
    const _Float16* Qb  = qh + b * NN * DD;
    const _Float16* Vtb = vt + b * DD * NN;
    const unsigned char* Mb = mask + (size_t)b * NN * NN;

    h16 bq0 = load_frag_b(Qb, n0, DD, 0,  li, hg);   // Q-tile fixed for this wave
    h16 bq1 = load_frag_b(Qb, n0, DD, 32, li, hg);

    f8 acc[4] = {};                                   // out tile: 16 n x 64 d

    for (int mt = 0; mt < NN / 32; ++mt) {
        int m0 = mt << 5;
        FragH pa;                                     // P A-fragment (16 n x 32 m), f16
        #pragma unroll
        for (int sub = 0; sub < 2; ++sub) {
            int ms = m0 + sub * 16;
            h16 aK0 = load_frag_a(Kb, ms, DD, 0,  li, hg);
            h16 aK1 = load_frag_a(Kb, ms, DD, 32, li, hg);
            f8 c = {0.f,0.f,0.f,0.f,0.f,0.f,0.f,0.f};
            c = wmma_f16(aK0, bq0, c);                // S^T tile
            c = wmma_f16(aK1, bq1, c);
            unsigned long long mb =
                *(const unsigned long long*)(Mb + (size_t)(n0 + li) * NN + ms + hg * 8);
            unsigned mlo = (unsigned)mb;
            unsigned mhi = (unsigned)(mb >> 32);
            // S^T C-fragment maps lane-locally onto the PV A-fragment (pack f32 pairs -> f16)
            #pragma unroll
            for (int j = 0; j < 4; ++j) {
                unsigned mw = (j < 2) ? mlo : mhi;
                unsigned b0 = (mw >> (16 * (j & 1)))     & 0xffu;
                unsigned b1 = (mw >> (16 * (j & 1) + 8)) & 0xffu;
                float e0 = __builtin_amdgcn_exp2f((float)c[2 * j]);
                float e1 = __builtin_amdgcn_exp2f((float)c[2 * j + 1]);
                e0 = b0 ? 0.f : e0;
                e1 = b1 ? 0.f : e1;
                auto p = __builtin_amdgcn_cvt_pkrtz(e0, e1);   // v_cvt_pk_rtz_f16_f32
                pa.w[sub * 4 + j] = __builtin_bit_cast(unsigned int, p);
            }
        }
        #pragma unroll
        for (int db = 0; db < 4; ++db) {
            h16 bv = load_frag_b(Vtb, db * 16, NN, m0, li, hg);   // Vt rows are d, K-dim = m
            acc[db] = wmma_f16(pa.h, bv, acc[db]);
        }
    }

    float* Ob = out + b * NN * DD;
    #pragma unroll
    for (int db = 0; db < 4; ++db) {
        #pragma unroll
        for (int r = 0; r < 8; ++r)
            Ob[(n0 + r + 8 * hg) * DD + db * 16 + li] = acc[db][r];
    }
}

extern "C" void kernel_launch(void* const* d_in, const int* in_sizes, int n_in,
                              void* d_out, int out_size, void* d_ws, size_t ws_size,
                              hipStream_t stream) {
    const float* q = (const float*)d_in[0];
    const float* k = (const float*)d_in[1];
    const float* v = (const float*)d_in[2];
    const unsigned char* mask = (const unsigned char*)d_in[3];
    float* out = (float*)d_out;

    char* ws = (char*)d_ws;
    _Float16* qh = (_Float16*)(ws);                             // 4 MB
    _Float16* kh = (_Float16*)(ws + (size_t)4  * 1024 * 1024);  // 4 MB
    _Float16* vt = (_Float16*)(ws + (size_t)8  * 1024 * 1024);  // 4 MB
    float*    iZ = (float*)  (ws + (size_t)12 * 1024 * 1024);   // 128 KB

    prep_qk<<<BB * NN * DD / 256, 256, 0, stream>>>(q, k, qh, kh);
    colsum_kernel<<<BB * NN / 128, 256, 0, stream>>>(qh, kh, mask, iZ);
    dim3 tb(32, 8);
    dim3 tg(NN / 32, DD / 32, BB);
    prep_vt<<<tg, tb, 0, stream>>>(v, iZ, vt);
    attn_out_kernel<<<BB * NN / 128, 256, 0, stream>>>(qh, kh, vt, mask, out);
}
